// DSConvSpectral_36180804501973
// MI455X (gfx1250) — matrix operand order repaired and checked by reading, not compiled, not run on hardware
//
#include <hip/hip_runtime.h>
#include <math.h>

#define BB 2
#define CC 64
#define HH 360
#define WW 720
#define NF 361
#define KW 23      // fwd K tiles over w: 23*32 = 736
#define JT 23      // fwd N tiles over j: 23*16 = 368
#define KJ 12      // inv K tiles over j: 12*32 = 384
#define WT 45      // inv N tiles over w: 45*16 = 720
#define SQRTN 26.832815729997478f

typedef __attribute__((ext_vector_type(16))) __bf16 v16bf;
typedef __attribute__((ext_vector_type(8)))  float  v8f;

__device__ __forceinline__ unsigned short f2bf(float f) {
    unsigned u = __float_as_uint(f);
    unsigned r = u + 0x7FFFu + ((u >> 16) & 1u);
    return (unsigned short)(r >> 16);
}
__device__ __forceinline__ float bf2f(unsigned short h) {
    return __uint_as_float(((unsigned)h) << 16);
}

// Load 16 bf16 (32B) as a WMMA operand without relying on 32B alignment of the base.
__device__ __forceinline__ v16bf ldop(const unsigned short* p) {
    union { uint4 q[2]; v16bf v; } u;
    u.q[0] = *(const uint4*)(p);
    u.q[1] = *(const uint4*)(p + 8);
    return u.v;
}

__device__ __forceinline__ v8f wmma_bf(const unsigned short* a, const unsigned short* b, v8f c) {
    return __builtin_amdgcn_wmma_f32_16x16x32_bf16(false, ldop(a), false, ldop(b),
                                                   (short)0, c, false, false);
}

// ---------------------------------------------------------------------------
// Table builders
// fwd tables: B-operand tiles [arr][kt][jt][lane][e], value = trig(2*pi*w*j/720)/sqrt(720)
//   w = kt*32 + e + 16*(lane>>4), j = jt*16 + (lane&15); zero for w>=720 or j>360
// ---------------------------------------------------------------------------
__global__ void k_build_fwd(unsigned short* __restrict__ fwd) {
    int idx = blockIdx.x * 256 + threadIdx.x;
    const int total = KW * JT * 512;
    if (idx >= total) return;
    int e = idx & 15, lane = (idx >> 4) & 31;
    int rest = idx >> 9;
    int jt = rest % JT, kt = rest / JT;
    int w = kt * 32 + e + 16 * (lane >> 4);
    int j = jt * 16 + (lane & 15);
    float c = 0.f, s = 0.f;
    if (w < WW && j < NF) {
        int m = (int)(((long long)w * j) % WW);
        float ang = 6.28318530717958647692f * (float)m / (float)WW;
        c = cosf(ang) / SQRTN;
        s = sinf(ang) / SQRTN;
    }
    unsigned short chi = f2bf(c); unsigned short clo = f2bf(c - bf2f(chi));
    unsigned short shi = f2bf(s); unsigned short slo = f2bf(s - bf2f(shi));
    fwd[idx] = chi; fwd[total + idx] = clo;
    fwd[2 * total + idx] = shi; fwd[3 * total + idx] = slo;
}

// inv tables: B-operand tiles [arr][kt][wt][lane][e], value = trig(2*pi*j*w/720)
//   j = kt*32 + e + 16*(lane>>4), w = wt*16 + (lane&15); zero for j>360
__global__ void k_build_inv(unsigned short* __restrict__ inv) {
    int idx = blockIdx.x * 256 + threadIdx.x;
    const int total = KJ * WT * 512;
    if (idx >= total) return;
    int e = idx & 15, lane = (idx >> 4) & 31;
    int rest = idx >> 9;
    int wt = rest % WT, kt = rest / WT;
    int j = kt * 32 + e + 16 * (lane >> 4);
    int w = wt * 16 + (lane & 15);
    float c = 0.f, s = 0.f;
    if (j < NF) {
        int m = (int)(((long long)j * w) % WW);
        float ang = 6.28318530717958647692f * (float)m / (float)WW;
        c = cosf(ang);
        s = sinf(ang);
    }
    unsigned short chi = f2bf(c); unsigned short clo = f2bf(c - bf2f(chi));
    unsigned short shi = f2bf(s); unsigned short slo = f2bf(s - bf2f(shi));
    inv[idx] = chi; inv[total + idx] = clo;
    inv[2 * total + idx] = shi; inv[3 * total + idx] = slo;
}

// Weight A-operand tiles: [arr6][ct4][kt2][lane][e]; arrays: w1r,w1i,-w1i,w2r,w2i,-w2i
__global__ void k_build_wts(const float* __restrict__ w1r, const float* __restrict__ w1i,
                            const float* __restrict__ w2r, const float* __restrict__ w2i,
                            unsigned short* __restrict__ wA) {
    int idx = blockIdx.x * 256 + threadIdx.x;
    if (idx >= 6 * 4096) return;
    int arr = idx >> 12;
    int r = idx & 4095;
    int e = r & 15, lane = (r >> 4) & 31, kt = (r >> 9) & 1, ct = r >> 10;
    int o = ct * 16 + (lane & 15);
    int i = kt * 32 + e + 16 * (lane >> 4);
    const float* src; float sgn = 1.f;
    switch (arr) {
        case 0: src = w1r; break;
        case 1: src = w1i; break;
        case 2: src = w1i; sgn = -1.f; break;
        case 3: src = w2r; break;
        case 4: src = w2i; break;
        default: src = w2i; sgn = -1.f; break;
    }
    wA[idx] = f2bf(sgn * src[o * CC + i]);
}

// ---------------------------------------------------------------------------
// Per-row mode counts, offsets (raw + padded-to-16), and tp -> (t, j) maps.
// meta layout: [0..359]=row_off, [360..719]=cnt, [720..1079]=prow_off, [1080]=TP
// ---------------------------------------------------------------------------
__global__ void k_offsets(const int* __restrict__ rows, int T,
                          int* __restrict__ meta, int* __restrict__ tmap,
                          int* __restrict__ jmap) {
    __shared__ int cnt[HH];
    __shared__ int roff[HH + 1], poff[HH + 1];
    int tid = threadIdx.x;
    for (int h = tid; h < HH; h += blockDim.x) cnt[h] = 0;
    __syncthreads();
    for (int t = tid; t < T; t += blockDim.x) atomicAdd(&cnt[rows[t]], 1);
    __syncthreads();
    if (tid == 0) {
        int a = 0, p = 0;
        for (int h = 0; h < HH; h++) {
            roff[h] = a; poff[h] = p;
            a += cnt[h];
            p += ((cnt[h] + 15) >> 4) << 4;
        }
        roff[HH] = a; poff[HH] = p;
    }
    __syncthreads();
    for (int h = tid; h < HH; h += blockDim.x) {
        meta[h] = roff[h];
        meta[HH + h] = cnt[h];
        meta[2 * HH + h] = poff[h];
    }
    if (tid == 0) meta[3 * HH] = poff[HH];
    __syncthreads();
    for (int h = tid; h < HH; h += blockDim.x) {
        int c = cnt[h], cp = ((c + 15) >> 4) << 4, r0 = roff[h], p0 = poff[h];
        for (int j = 0; j < cp; j++) {
            tmap[p0 + j] = r0 + (j < c ? j : (c - 1));
            jmap[p0 + j] = j;
        }
    }
}

// ---------------------------------------------------------------------------
// Forward DFT per (b,h) + normalization.  LDS: x in A-layout, bf16 hi+lo.
// Outputs: hn (bf16, B-operand layout per 16-mode tile), xm (f32, C layout).
// ---------------------------------------------------------------------------
__global__ void __launch_bounds__(128) k_fwd(
    const float* __restrict__ x,
    const float* __restrict__ smr, const float* __restrict__ smi,
    const float* __restrict__ sstd,
    const float* __restrict__ magr, const float* __restrict__ magi,
    const float* __restrict__ bir, const float* __restrict__ bii,
    const unsigned short* __restrict__ fwdT,
    const int* __restrict__ meta, int TTmax,
    unsigned short* __restrict__ hnr, unsigned short* __restrict__ hni,
    float* __restrict__ xmr, float* __restrict__ xmi) {
    extern __shared__ unsigned short smA[];
    unsigned short* xhi = smA;                 // 4*23*512
    unsigned short* xlo = smA + 4 * KW * 512;  // 4*23*512
    const int b = blockIdx.x / HH, h = blockIdx.x % HH;
    const int tid = threadIdx.x;

    for (int idx = tid; idx < CC * 736; idx += 128) {
        int c = idx / 736, w = idx % 736;
        float v = (w < WW) ? x[(((size_t)b * CC + c) * HH + h) * WW + w] : 0.f;
        unsigned short vh = f2bf(v);
        unsigned short vl = f2bf(v - bf2f(vh));
        int ct = c >> 4, ll = c & 15, kt = w >> 5, r = w & 31;
        int half = r >> 4, e = r & 15;
        int lane = ll + 16 * half;
        int o = ((ct * KW + kt) * 32 + lane) * 16 + e;
        xhi[o] = vh; xlo[o] = vl;
    }
    __syncthreads();

    const int roff = meta[h], cn = meta[HH + h], poff = meta[2 * HH + h];
    const int jtn = (cn + 15) >> 4;
    const int wave = tid >> 5, lane = tid & 31;
    const int ct = wave;
    const int n = lane & 15, half = lane >> 4;
    const size_t ATO = (size_t)KW * JT * 512;

    for (int jt = 0; jt < jtn; jt++) {
        v8f accC = {}; v8f accS = {};
        for (int kt = 0; kt < KW; kt++) {
            const unsigned short* ah = &xhi[((ct * KW + kt) * 32 + lane) * 16];
            const unsigned short* al = &xlo[((ct * KW + kt) * 32 + lane) * 16];
            size_t bo = ((size_t)(kt * JT + jt) * 32 + lane) * 16;
            const unsigned short* bch = fwdT + bo;
            const unsigned short* bcl = fwdT + ATO + bo;
            const unsigned short* bsh = fwdT + 2 * ATO + bo;
            const unsigned short* bsl = fwdT + 3 * ATO + bo;
            accC = wmma_bf(ah, bch, accC);   // bf16x3 split: hi*hi + hi*lo + lo*hi
            accC = wmma_bf(ah, bcl, accC);
            accC = wmma_bf(al, bch, accC);
            accS = wmma_bf(ah, bsh, accS);
            accS = wmma_bf(ah, bsl, accS);
            accS = wmma_bf(al, bsh, accS);
        }
        int jv = jt * 16 + n;
        bool valid = jv < cn;
        int t = roff + (valid ? jv : 0);
        float mr_t = smr[t], mi_t = smi[t];
        float istd = 1.f / (1e-12f + sstd[t]);
        int tp = poff + jt * 16 + n;
        int tt = tp >> 4;
        for (int v = 0; v < 8; v++) {
            int c = ct * 16 + v + 8 * half;
            float Xr = accC[v], Xi = -accS[v];   // rfft: X = sum x (cos - i sin)
            float xr = 0.f, xi = 0.f, hr = 0.f, hq = 0.f;
            if (valid) {
                xr = Xr; xi = Xi;
                float ur = (Xr - mr_t) * istd, ui = (Xi - mi_t) * istd;
                float Mr = magr[c], Mi = magi[c];
                hr = ur * Mr - ui * Mi + bir[c];
                hq = ur * Mi + ui * Mr + bii[c];
            }
            size_t xidx = ((((size_t)b * TTmax + tt) * 4 + ct) * 32 + lane) * 8 + v;
            xmr[xidx] = xr; xmi[xidx] = xi;
            int kti = c >> 5, rr = c & 31, hb = rr >> 4, eB = rr & 15;
            int laneB = n + 16 * hb;
            size_t hidx = ((((size_t)b * TTmax + tt) * 2 + kti) * 32 + laneB) * 16 + eB;
            hnr[hidx] = f2bf(hr); hni[hidx] = f2bf(hq);
        }
    }
}

// ---------------------------------------------------------------------------
// Channel mixer per (b, 16-mode tile): cgemm1 -> |.|-GELU -> cgemm2 -> GLU gate.
// Emits gm packed bf16 (real, -imag) scaled by 2/sqrt(N)*alpha_j.
// ---------------------------------------------------------------------------
__global__ void __launch_bounds__(128) k_mix(
    const unsigned short* __restrict__ wA,
    const unsigned short* __restrict__ hnr, const unsigned short* __restrict__ hni,
    const float* __restrict__ xmr, const float* __restrict__ xmi,
    const float* __restrict__ glum, const float* __restrict__ glup,
    const float* __restrict__ brelu,
    const int* __restrict__ meta, const int* __restrict__ tmap,
    const int* __restrict__ jmap, int TTmax, int T,
    unsigned int* __restrict__ gm) {
    __shared__ __attribute__((aligned(32))) unsigned short h2r[2 * 512];
    __shared__ __attribute__((aligned(32))) unsigned short h2i[2 * 512];
    const int TP = meta[3 * HH];
    const int b = blockIdx.x / TTmax, tt = blockIdx.x % TTmax;
    if (tt * 16 >= TP) return;
    const int tid = threadIdx.x, wave = tid >> 5, lane = tid & 31, ct = wave;
    const int n = lane & 15, half = lane >> 4;

    v8f ar = {}, ai = {};
    for (int kt = 0; kt < 2; kt++) {
        const unsigned short* Awr = wA + (size_t)((0 * 4 + ct) * 2 + kt) * 512 + lane * 16;
        const unsigned short* Awi = wA + (size_t)((1 * 4 + ct) * 2 + kt) * 512 + lane * 16;
        const unsigned short* Awn = wA + (size_t)((2 * 4 + ct) * 2 + kt) * 512 + lane * 16;
        size_t bo = ((((size_t)b * TTmax + tt) * 2 + kt) * 32 + lane) * 16;
        const unsigned short* Br = hnr + bo;
        const unsigned short* Bi = hni + bo;
        ar = wmma_bf(Awr, Br, ar);   // Re: Wr*Hr + (-Wi)*Hi
        ar = wmma_bf(Awn, Bi, ar);
        ai = wmma_bf(Awr, Bi, ai);   // Im: Wr*Hi + Wi*Hr
        ai = wmma_bf(Awi, Br, ai);
    }
    const float br = brelu[0];
    for (int v = 0; v < 8; v++) {
        int c = ct * 16 + v + 8 * half;
        float hr = ar[v], hq = ai[v];
        float mag = sqrtf(hr * hr + hq * hq);
        float m2 = mag + br;
        float g = 0.5f * m2 * (1.f + erff(m2 * 0.70710678f));
        float s = g / fmaxf(mag, 1e-30f);
        int kti = c >> 5, rr = c & 31, hb = rr >> 4, eB = rr & 15;
        int o = ((kti * 32) + (n + 16 * hb)) * 16 + eB;
        h2r[o] = f2bf(hr * s);
        h2i[o] = f2bf(hq * s);
    }
    __syncthreads();

    v8f cr = {}, ci = {};
    for (int kt = 0; kt < 2; kt++) {
        const unsigned short* Awr = wA + (size_t)((3 * 4 + ct) * 2 + kt) * 512 + lane * 16;
        const unsigned short* Awi = wA + (size_t)((4 * 4 + ct) * 2 + kt) * 512 + lane * 16;
        const unsigned short* Awn = wA + (size_t)((5 * 4 + ct) * 2 + kt) * 512 + lane * 16;
        const unsigned short* Br = h2r + kt * 512 + lane * 16;
        const unsigned short* Bi = h2i + kt * 512 + lane * 16;
        cr = wmma_bf(Awr, Br, cr);
        cr = wmma_bf(Awn, Bi, cr);
        ci = wmma_bf(Awr, Bi, ci);
        ci = wmma_bf(Awi, Br, ci);
    }
    const int tp = tt * 16 + n;
    const int t = tmap[tp], jv = jmap[tp];
    const float scale = (2.f / SQRTN) * ((jv == 0 || jv == 360) ? 0.5f : 1.f);
    for (int v = 0; v < 8; v++) {
        int c = ct * 16 + v + 8 * half;
        float hr = cr[v], hq = ci[v];
        float mag = sqrtf(hr * hr + hq * hq);
        float sig = 1.f / (1.f + expf(-(mag + glum[(size_t)c * T + t])));
        float sp, cp;
        __sincosf(glup[(size_t)c * T + t], &sp, &cp);
        float inv = 1.f / fmaxf(mag, 1e-30f);
        float dr = (hr * cp - hq * sp) * inv;   // exp(i(angle(h)+phi))
        float di = (hr * sp + hq * cp) * inv;
        float gr = sig * dr, gi = sig * di;
        size_t xidx = ((((size_t)b * TTmax + tt) * 4 + ct) * 32 + lane) * 8 + v;
        float xr = xmr[xidx], xi = xmi[xidx];
        float pr = (xr * gr - xi * gi) * scale;
        float pn = -(xr * gi + xi * gr) * scale;   // store -imag for inverse GEMM
        gm[xidx] = (unsigned)f2bf(pr) | ((unsigned)f2bf(pn) << 16);
    }
}

// ---------------------------------------------------------------------------
// Inverse DFT per (b,h): gm staged to LDS A-layout, y = gr*cos + (-gi)*sin.
// ---------------------------------------------------------------------------
__global__ void __launch_bounds__(128) k_inv(
    const unsigned int* __restrict__ gm,
    const unsigned short* __restrict__ invT,
    const int* __restrict__ meta, int TTmax,
    float* __restrict__ y) {
    extern __shared__ unsigned short smI[];
    unsigned short* gr = smI;                 // 4*12*512
    unsigned short* gi = smI + 4 * KJ * 512;  // 4*12*512
    const int b = blockIdx.x / HH, h = blockIdx.x % HH;
    const int tid = threadIdx.x;
    const int cn = meta[HH + h], poff = meta[2 * HH + h];
    const int cpad = ((cn + 15) >> 4) << 4;

    for (int idx = tid; idx < CC * KJ * 32; idx += 128) {   // 64 * 384
        int c = idx / (KJ * 32), j = idx % (KJ * 32);
        unsigned pv = 0;
        if (j < cpad) {
            int tp = poff + j, tt = tp >> 4, nn = tp & 15;
            int ct0 = c >> 4, m = c & 15, hf = m >> 3, v0 = m & 7;
            int ln = nn + 16 * hf;
            pv = gm[((((size_t)b * TTmax + tt) * 4 + ct0) * 32 + ln) * 8 + v0];
        }
        int ct = c >> 4, ll = c & 15, kt = j >> 5, r = j & 31;
        int half = r >> 4, e = r & 15;
        int lane = ll + 16 * half;
        int o = ((ct * KJ + kt) * 32 + lane) * 16 + e;
        gr[o] = (unsigned short)(pv & 0xFFFFu);
        gi[o] = (unsigned short)(pv >> 16);
    }
    __syncthreads();

    const int wave = tid >> 5, lane = tid & 31, ct = wave;
    const int n = lane & 15, half = lane >> 4;
    const int ktn = (cpad + 31) >> 5;
    const size_t ATO = (size_t)KJ * WT * 512;
    for (int wt = 0; wt < WT; wt++) {
        v8f acc = {};
        for (int kt = 0; kt < ktn; kt++) {
            const unsigned short* Ar = &gr[((ct * KJ + kt) * 32 + lane) * 16];
            const unsigned short* Ai = &gi[((ct * KJ + kt) * 32 + lane) * 16];
            size_t bo = ((size_t)(kt * WT + wt) * 32 + lane) * 16;
            acc = wmma_bf(Ar, invT + bo, acc);            // gr * cos_hi
            acc = wmma_bf(Ar, invT + ATO + bo, acc);      // gr * cos_lo
            acc = wmma_bf(Ai, invT + 2 * ATO + bo, acc);  // (-gi) * sin_hi
            acc = wmma_bf(Ai, invT + 3 * ATO + bo, acc);  // (-gi) * sin_lo
        }
        for (int v = 0; v < 8; v++) {
            int c = ct * 16 + v + 8 * half;
            y[(((size_t)b * CC + c) * HH + h) * WW + wt * 16 + n] = acc[v];
        }
    }
}

// ---------------------------------------------------------------------------
// Depthwise 11-tap conv over latitude with great-circle wrap; out = y + z.
// ---------------------------------------------------------------------------
__global__ void __launch_bounds__(256) k_conv(
    const float* __restrict__ y, const float* __restrict__ dw,
    float* __restrict__ out) {
    int blk = blockIdx.x;
    int h = blk % HH;
    int bc = blk / HH;
    int c = bc % CC, b = bc / CC;
    float wk[11];
#pragma unroll
    for (int k = 0; k < 11; k++) wk[k] = dw[c * 11 + k];
    const float* yb = y + ((size_t)b * CC + c) * HH * WW;
    for (int w = threadIdx.x; w < WW; w += 256) {
        float acc = yb[(size_t)h * WW + w];
        float z = 0.f;
#pragma unroll
        for (int k = 0; k < 11; k++) {
            int r = h + k;
            int sh, wrap;
            if (r < 5)        { sh = 4 - r;   wrap = 1; }
            else if (r < 365) { sh = r - 5;   wrap = 0; }
            else              { sh = 724 - r; wrap = 1; }
            int ww = wrap ? ((w + 360) % 720) : w;
            z += wk[k] * yb[(size_t)sh * WW + ww];
        }
        out[(((size_t)b * CC + c) * HH + h) * WW + w] = acc + z;
    }
}

// ---------------------------------------------------------------------------
extern "C" void kernel_launch(void* const* d_in, const int* in_sizes, int n_in,
                              void* d_out, int out_size, void* d_ws, size_t ws_size,
                              hipStream_t stream) {
    (void)n_in; (void)out_size;
    const float* x    = (const float*)d_in[0];
    const float* smr  = (const float*)d_in[1];
    const float* smi  = (const float*)d_in[2];
    const float* sstd = (const float*)d_in[3];
    const float* magr = (const float*)d_in[4];
    const float* magi = (const float*)d_in[5];
    const float* bir  = (const float*)d_in[6];
    const float* bii  = (const float*)d_in[7];
    const float* w1r  = (const float*)d_in[8];
    const float* w1i  = (const float*)d_in[9];
    const float* brel = (const float*)d_in[10];
    const float* w2r  = (const float*)d_in[11];
    const float* w2i  = (const float*)d_in[12];
    const float* glum = (const float*)d_in[13];
    const float* glup = (const float*)d_in[14];
    const float* dw   = (const float*)d_in[15];
    const int*   rows = (const int*)d_in[16];
    const int T = in_sizes[16];

    const int TPmax = ((T + HH * 16) + 15) & ~15;
    const int TTmax = TPmax >> 4;

    char* p = (char*)d_ws;
    auto alloc = [&](size_t sz) -> char* {
        char* r = p;
        p += (sz + 255) & ~(size_t)255;
        return r;
    };
    unsigned short* fwdT = (unsigned short*)alloc((size_t)4 * KW * JT * 512 * 2);
    unsigned short* invT = (unsigned short*)alloc((size_t)4 * KJ * WT * 512 * 2);
    unsigned short* wA   = (unsigned short*)alloc((size_t)6 * 4096 * 2);
    int* meta            = (int*)alloc((size_t)(3 * HH + 8) * 4);
    int* tmap            = (int*)alloc((size_t)TPmax * 4);
    int* jmap            = (int*)alloc((size_t)TPmax * 4);
    unsigned short* hnr  = (unsigned short*)alloc((size_t)BB * TTmax * 1024 * 2);
    unsigned short* hni  = (unsigned short*)alloc((size_t)BB * TTmax * 1024 * 2);
    float* xmr           = (float*)alloc((size_t)BB * TTmax * 1024 * 4);
    float* xmi           = (float*)alloc((size_t)BB * TTmax * 1024 * 4);
    unsigned int* gmb    = (unsigned int*)alloc((size_t)BB * TTmax * 1024 * 4);
    float* ybuf          = (float*)alloc((size_t)BB * CC * HH * WW * 4);
    if ((size_t)(p - (char*)d_ws) > ws_size) return;

    const int ldsFwd = 2 * 4 * KW * 512 * 2;   // 188,416 B
    const int ldsInv = 2 * 4 * KJ * 512 * 2;   //  98,304 B
    hipFuncSetAttribute(reinterpret_cast<const void*>(k_fwd),
                        hipFuncAttributeMaxDynamicSharedMemorySize, ldsFwd);
    hipFuncSetAttribute(reinterpret_cast<const void*>(k_inv),
                        hipFuncAttributeMaxDynamicSharedMemorySize, ldsInv);

    k_build_fwd<<<(KW * JT * 512 + 255) / 256, 256, 0, stream>>>(fwdT);
    k_build_inv<<<(KJ * WT * 512 + 255) / 256, 256, 0, stream>>>(invT);
    k_build_wts<<<(6 * 4096 + 255) / 256, 256, 0, stream>>>(w1r, w1i, w2r, w2i, wA);
    k_offsets<<<1, 512, 0, stream>>>(rows, T, meta, tmap, jmap);
    k_fwd<<<BB * HH, 128, ldsFwd, stream>>>(x, smr, smi, sstd, magr, magi, bir, bii,
                                            fwdT, meta, TTmax, hnr, hni, xmr, xmi);
    k_mix<<<BB * TTmax, 128, 0, stream>>>(wA, hnr, hni, xmr, xmi, glum, glup, brel,
                                          meta, tmap, jmap, TTmax, T, gmb);
    k_inv<<<BB * HH, 128, ldsInv, stream>>>(gmb, invT, meta, TTmax, ybuf);
    k_conv<<<BB * CC * HH, 256, 0, stream>>>(ybuf, dw, (float*)d_out);
}
